// VisionAttention_34703335752050
// MI455X (gfx1250) — matrix operand order repaired and checked by reading, not compiled, not run on hardware
//
#include <hip/hip_runtime.h>

// ---------------- constants ----------------
constexpr int kSEQ = 4096;
constexpr int kHID = 1280;
constexpr int kNH  = 16;
constexpr int kHD  = 80;    // head dim
constexpr int kHDP = 96;    // head dim padded to 3*32 for WMMA K-chains

typedef __attribute__((ext_vector_type(16))) __bf16 v16bf;
typedef __attribute__((ext_vector_type(8)))  float  v8f;

union FragU { uint4 u[2]; v16bf v; };

// ---- bf16 16x16x32 fragment loaders (wave32 layouts per CDNA5 ISA 7.12.2) ----
// A (16x32, MxK): lane = M%16 | half; per-half K map: e<8 -> k0+e+half*8,
//                 e>=8 -> k0+16+(e-8)+half*8   (two 16B loads)
__device__ inline v16bf load_frag_A(const __bf16* base, int ld, int row0, int k0, int lane) {
  int m = lane & 15, hf = lane >> 4;
  const __bf16* p = base + (size_t)(row0 + m) * ld + (k0 + hf * 8);
  FragU f;
  f.u[0] = *(const uint4*)(p);
  f.u[1] = *(const uint4*)(p + 16);
  return f.v;
}

// B (32x16, KxN) where storage is row-major [N][K] (weight[n][k] or k-major kv):
// lane = N%16 | half; K map: e<8 -> k0+e+half*8, e>=8 -> k0+16+(e-8)+half*8  (two 16B loads)
__device__ inline v16bf load_frag_Bnk(const __bf16* base, long ld, int n0, int k0, int lane) {
  int n = lane & 15, hf = lane >> 4;
  const __bf16* p = base + (size_t)(n0 + n) * ld + (k0 + hf * 8);
  FragU f;
  f.u[0] = *(const uint4*)(p);
  f.u[1] = *(const uint4*)(p + 16);
  return f.v;
}

__device__ inline v8f wmma_bf16(v16bf a, v16bf b, v8f c) {
  return __builtin_amdgcn_wmma_f32_16x16x32_bf16(false, a, false, b, (short)0, c, false, false);
}

// ---------------- kernel 0: fp32 -> bf16 ----------------
__global__ void cvt_bf16_kernel(const float* __restrict__ in, __bf16* __restrict__ out, int n) {
  int i = blockIdx.x * blockDim.x + threadIdx.x;
  int stride = gridDim.x * blockDim.x;
  for (; i < n; i += stride) out[i] = (__bf16)in[i];
}

// ---------------- generic GEMM: out[M,N] = A[M,K] @ W[N,K]^T + bias ----------------
// one wave (32 threads) per 32x32 output block (2x2 register-blocked WMMA tiles):
// each A/B fragment is reused by 2 WMMAs -> 16 FLOP/byte of fragment traffic,
// and 4 independent WMMAs per k-step hide load latency.
__global__ void gemm_bf16_kernel(const __bf16* __restrict__ A, const __bf16* __restrict__ W,
                                 const float* __restrict__ bias, float* __restrict__ out,
                                 int M, int N, int K) {
  int lane = threadIdx.x;
  int n0 = blockIdx.x * 32;
  int m0 = blockIdx.y * 32;
  v8f c00 = {}, c01 = {}, c10 = {}, c11 = {};
#pragma unroll 2
  for (int k0 = 0; k0 < K; k0 += 32) {
    v16bf a0 = load_frag_A(A, K, m0,      k0, lane);
    v16bf a1 = load_frag_A(A, K, m0 + 16, k0, lane);
    v16bf b0 = load_frag_Bnk(W, K, n0,      k0, lane);
    v16bf b1 = load_frag_Bnk(W, K, n0 + 16, k0, lane);
    c00 = wmma_bf16(a0, b0, c00);
    c01 = wmma_bf16(a0, b1, c01);
    c10 = wmma_bf16(a1, b0, c10);
    c11 = wmma_bf16(a1, b1, c11);
  }
  int hf = lane >> 4, ln = lane & 15;
  int col0 = n0 + ln, col1 = n0 + 16 + ln;
  float bv0 = bias[col0];
  float bv1 = bias[col1];
#pragma unroll
  for (int j = 0; j < 8; ++j) {
    int row0 = m0 + j + 8 * hf;       // C/D layout: vgpr j holds M = j + 8*half
    int row1 = row0 + 16;
    out[(size_t)row0 * N + col0] = c00[j] + bv0;
    out[(size_t)row0 * N + col1] = c01[j] + bv1;
    out[(size_t)row1 * N + col0] = c10[j] + bv0;
    out[(size_t)row1 * N + col1] = c11[j] + bv1;
  }
}

// ---------------- RoPE + repack ----------------
// qkvf: [SEQ][3*HID] f32.  Produces:
//   qh/kh: [NH][SEQ][HDP] bf16 (zero-padded K, softmax scale folded into q)
//   vt:    [NH][HD][SEQ]  bf16 (transposed so PV B-operand is K-contiguous)
__global__ void rope_pack_kernel(const float* __restrict__ qkvf,
                                 const float* __restrict__ cosb, const float* __restrict__ sinb,
                                 __bf16* __restrict__ qh, __bf16* __restrict__ kh,
                                 __bf16* __restrict__ vt) {
  int s = blockIdx.x;
  int h = blockIdx.y;
  int d = threadIdx.x;           // 0..95
  size_t qo = ((size_t)h * kSEQ + s) * kHDP + d;
  if (d < kHD) {
    float c  = cosb[(size_t)s * kHD + d];
    float sn = sinb[(size_t)s * kHD + d];
    const float* base = qkvf + (size_t)s * 3 * kHID + h * kHD;
    float qv = base[d];
    float kv = base[kHID + d];
    float vv = base[2 * kHID + d];
    int   d2  = (d < kHD / 2) ? d + kHD / 2 : d - kHD / 2;
    float sgn = (d < kHD / 2) ? -1.0f : 1.0f;
    float qr = sgn * base[d2];
    float kr = sgn * base[kHID + d2];
    float q = qv * c + qr * sn;
    float k = kv * c + kr * sn;
    qh[qo] = (__bf16)(q * 0.11180339887498949f);   // 1/sqrt(80) folded in
    kh[qo] = (__bf16)k;
    vt[((size_t)h * kHD + d) * kSEQ + s] = (__bf16)vv;
  } else {
    qh[qo] = (__bf16)0.0f;
    kh[qo] = (__bf16)0.0f;
  }
}

// ---------------- flash attention: one wave per (head, 16-row q tile) ----------------
__global__ void attn_kernel(const __bf16* __restrict__ qh, const __bf16* __restrict__ kh,
                            const __bf16* __restrict__ vt, const int* __restrict__ cu,
                            __bf16* __restrict__ attn_out) {
  __shared__ __bf16 Pst[16][32];          // C-layout -> A-layout relayout buffer (1 KB)
  int lane = threadIdx.x;
  int hf = lane >> 4, ln = lane & 15;
  int q0 = blockIdx.x * 16;
  int h  = blockIdx.y;

  // segment bounds (block-diagonal bias); tiles are segment-aligned
  int beg = 0, end = kSEQ;
  for (int i = 0; i < 4; ++i) {
    int a = cu[i], b = cu[i + 1];
    if (q0 >= a && q0 < b) { beg = a; end = b; }
  }

  const __bf16* Q  = qh + (size_t)h * kSEQ * kHDP;
  const __bf16* Kp = kh + (size_t)h * kSEQ * kHDP;
  const __bf16* V  = vt + (size_t)h * kHD * kSEQ;

  v16bf aq[3];
#pragma unroll
  for (int i = 0; i < 3; ++i) aq[i] = load_frag_A(Q, kHDP, q0, i * 32, lane);

  v8f o[5];
  float m[8], l[8];
#pragma unroll
  for (int nf = 0; nf < 5; ++nf) o[nf] = v8f{};
#pragma unroll
  for (int j = 0; j < 8; ++j) { m[j] = -1e30f; l[j] = 0.0f; }

  for (int key0 = beg; key0 < end; key0 += 32) {
    // scores for 16 rows x 32 keys: two 16x16 tiles, K-chain of 3 (HDP=96)
    v8f s0 = {}, s1 = {};
#pragma unroll
    for (int kk = 0; kk < 3; ++kk) {
      v16bf b0 = load_frag_Bnk(Kp, kHDP, key0,      kk * 32, lane);
      v16bf b1 = load_frag_Bnk(Kp, kHDP, key0 + 16, kk * 32, lane);
      s0 = wmma_bf16(aq[kk], b0, s0);
      s1 = wmma_bf16(aq[kk], b1, s1);
    }
    // online softmax; row r = j + 8*half, cols striped over 16 lanes of the half
#pragma unroll
    for (int j = 0; j < 8; ++j) {
      float v0 = s0[j], v1 = s1[j];
      float mx = fmaxf(v0, v1);
#pragma unroll
      for (int off = 8; off > 0; off >>= 1) mx = fmaxf(mx, __shfl_xor(mx, off, 32));
      float mn = fmaxf(m[j], mx);
      float sc = __expf(m[j] - mn);
      float p0 = __expf(v0 - mn);
      float p1 = __expf(v1 - mn);
      float rs = p0 + p1;
#pragma unroll
      for (int off = 8; off > 0; off >>= 1) rs += __shfl_xor(rs, off, 32);
      l[j] = l[j] * sc + rs;
      m[j] = mn;
#pragma unroll
      for (int nf = 0; nf < 5; ++nf) o[nf][j] *= sc;
      int row = j + 8 * hf;
      Pst[row][ln]      = (__bf16)p0;
      Pst[row][ln + 16] = (__bf16)p1;
    }
    // P (16x32 bf16, A-layout from LDS) @ V (32x80): 5 n-tiles
    v16bf ap = load_frag_A(&Pst[0][0], 32, 0, 0, lane);
#pragma unroll
    for (int nf = 0; nf < 5; ++nf) {
      v16bf bv = load_frag_Bnk(V, kSEQ, nf * 16, key0, lane);
      o[nf] = wmma_bf16(ap, bv, o[nf]);
    }
  }

  // epilogue: normalize and scatter to [SEQ][HID] bf16
#pragma unroll
  for (int j = 0; j < 8; ++j) {
    float inv = 1.0f / l[j];
    int row = q0 + j + 8 * hf;
#pragma unroll
    for (int nf = 0; nf < 5; ++nf) {
      int col = h * kHD + nf * 16 + ln;
      attn_out[(size_t)row * kHID + col] = (__bf16)(o[nf][j] * inv);
    }
  }
}

extern "C" void kernel_launch(void* const* d_in, const int* in_sizes, int n_in,
                              void* d_out, int out_size, void* d_ws, size_t ws_size,
                              hipStream_t stream) {
  const float* hidden = (const float*)d_in[0];
  const float* cosb   = (const float*)d_in[1];
  const float* sinb   = (const float*)d_in[2];
  const float* qkv_w  = (const float*)d_in[3];
  const float* qkv_b  = (const float*)d_in[4];
  const float* proj_w = (const float*)d_in[5];
  const float* proj_b = (const float*)d_in[6];
  const int*   cu     = (const int*)d_in[7];
  float* out = (float*)d_out;

  char* p = (char*)d_ws;
  auto take = [&p](size_t bytes) -> char* {
    char* r = p;
    p += (bytes + 255) & ~(size_t)255;
    return r;
  };
  __bf16* hiddenB = (__bf16*)take((size_t)kSEQ * kHID * 2);
  __bf16* qkvwB   = (__bf16*)take((size_t)3 * kHID * kHID * 2);
  __bf16* projwB  = (__bf16*)take((size_t)kHID * kHID * 2);
  float*  qkvf    = (float*) take((size_t)kSEQ * 3 * kHID * 4);
  __bf16* qh      = (__bf16*)take((size_t)kNH * kSEQ * kHDP * 2);
  __bf16* kh      = (__bf16*)take((size_t)kNH * kSEQ * kHDP * 2);
  __bf16* vt      = (__bf16*)take((size_t)kNH * kHD * kSEQ * 2);
  __bf16* attnB   = (__bf16*)take((size_t)kSEQ * kHID * 2);

  // 0) convert operands to bf16 (L2-resident afterwards: ~24 MB total)
  cvt_bf16_kernel<<<4096, 256, 0, stream>>>(hidden, hiddenB, kSEQ * kHID);
  cvt_bf16_kernel<<<4096, 256, 0, stream>>>(qkv_w, qkvwB, 3 * kHID * kHID);
  cvt_bf16_kernel<<<4096, 256, 0, stream>>>(proj_w, projwB, kHID * kHID);

  // 1) QKV = hidden @ qkv_w^T + qkv_b   (fp32 accumulate via bf16 WMMA)
  gemm_bf16_kernel<<<dim3(3 * kHID / 32, kSEQ / 32), 32, 0, stream>>>(
      hiddenB, qkvwB, qkv_b, qkvf, kSEQ, 3 * kHID, kHID);

  // 2) RoPE + head-major repack (+ v transpose, + softmax scale fold)
  rope_pack_kernel<<<dim3(kSEQ, kNH), kHDP, 0, stream>>>(qkvf, cosb, sinb, qh, kh, vt);

  // 3) block-diagonal flash attention
  attn_kernel<<<dim3(kSEQ / 16, kNH), 32, 0, stream>>>(qh, kh, vt, cu, attnB);

  // 4) out = attn @ proj_w^T + proj_b (fp32 out)
  gemm_bf16_kernel<<<dim3(kHID / 32, kSEQ / 32), 32, 0, stream>>>(
      attnB, projwB, proj_b, out, kSEQ, kHID, kHID);
}
